// SpatialLayer_60636348285728
// MI455X (gfx1250) — matrix operand order repaired and checked by reading, not compiled, not run on hardware
//
#include <hip/hip_runtime.h>
#include <hip/hip_bf16.h>
#include <math.h>

typedef __attribute__((ext_vector_type(16))) _Float16 v16h;
typedef __attribute__((ext_vector_type(8)))  float    v8f;
typedef __attribute__((ext_vector_type(4)))  float    f4;

#define JN   25
#define BN   24
#define NB   4
#define CIN  64
#define COUT 64
#define TT   64
#define NP   625          // J*J
#define NPAD 640          // padded to 40 tiles of 16
#define KDIM 192

// -------- LDS (all operand tiles stored K-contiguous / transposed) --------
__shared__ _Float16 sJT[JN][CIN];      // joints^T:  [joint][channel]
__shared__ _Float16 sBT[BN][CIN];      // bones^T:   [bone][channel]
__shared__ _Float16 sPathT[NP][CIN];   // path-means^T: [pair p][channel] (bpf then jpf)
__shared__ float    sLogits[NPAD];
__shared__ float    sAttn[NPAD];
__shared__ float    sJOutT[JN][COUT];  // joints_out^T accumulator [joint][channel]
__shared__ int      sE0[BN], sE1[BN];

__device__ __forceinline__ float gelu_exact(float x) {
    return 0.5f * x * (1.0f + erff(x * 0.70710678118654752f));
}

// 16x32 f16 A-fragment of weight W (row-major [64][192]) for tile (mt,kt).
// Lane layout: m = mt*16 + (lane&15); halves q:
//   q 0..15 -> k = kt*32 + hi*16 + q
// i.e. one contiguous float16 run -> 4x float4 global loads + cvt.
__device__ __forceinline__ v16h load_a_frag(const float* W, int mt, int kt,
                                            int nl, int hi) {
    const float* base = W + (mt * 16 + nl) * KDIM + kt * 32 + hi * 16;
    f4 w0 = *(const f4*)(base);
    f4 w1 = *(const f4*)(base + 4);
    f4 w2 = *(const f4*)(base + 8);
    f4 w3 = *(const f4*)(base + 12);
    v16h a;
#pragma unroll
    for (int q = 0; q < 4; ++q) {
        a[q]      = (_Float16)w0[q];
        a[q + 4]  = (_Float16)w1[q];
        a[q + 8]  = (_Float16)w2[q];
        a[q + 12] = (_Float16)w3[q];
    }
    return a;
}

// Path means (transposed): sPathT[p][c] = sum_l mask*src[paths[l]][c] / len[p]
// Wave handles p strided by 8; lanes cover channels c and c+32 (conflict-free rows).
__device__ __forceinline__ void path_means(int wave, int lane,
                                           const int* paths, const float* mask,
                                           const float* len, int L,
                                           const _Float16 (*srcT)[CIN]) {
    for (int p = wave; p < NP; p += 8) {
        float a0 = 0.f, a1 = 0.f;
        for (int l = 0; l < L; ++l) {
            float m  = mask[p * L + l];
            int  idx = paths[p * L + l];
            a0 += m * (float)srcT[idx][lane];
            a1 += m * (float)srcT[idx][lane + 32];
        }
        float inv = 1.0f / len[p];
        sPathT[p][lane]      = (_Float16)(a0 * inv);
        sPathT[p][lane + 32] = (_Float16)(a1 * inv);
    }
}

__global__ void __launch_bounds__(256)
spatial_layer_fused(const float* __restrict__ joints,
                    const float* __restrict__ bones,
                    const float* __restrict__ W1, const float* __restrict__ b1,
                    const float* __restrict__ W2,
                    const float* __restrict__ Wt, const float* __restrict__ bt,
                    const float* __restrict__ Wb, const float* __restrict__ bb,
                    const int*   __restrict__ edges,
                    const int*   __restrict__ jpaths, const float* __restrict__ jmask,
                    const float* __restrict__ jlen,
                    const int*   __restrict__ bpaths, const float* __restrict__ bmask,
                    const float* __restrict__ blen,
                    float* __restrict__ out, int Lj, int Lb) {
    const int tid  = threadIdx.x;
    const int lane = tid & 31;
    const int wave = tid >> 5;
    const int hi   = lane >> 4;       // half-wave select
    const int nl   = lane & 15;       // lane within half
    const int nIdx = blockIdx.x >> 6; // batch
    const int t    = blockIdx.x & 63; // time step

    // ---------------- Phase A: stage tiles (transposed), zero accumulators ---
    for (int idx = tid; idx < CIN * JN; idx += 256) {
        int c = idx / JN, j = idx - c * JN;   // global reads coalesced along j
        sJT[j][c] = (_Float16)joints[(((size_t)(nIdx * CIN + c) * TT) + t) * JN + j];
    }
    for (int idx = tid; idx < CIN * BN; idx += 256) {
        int c = idx / BN, j = idx - c * BN;
        sBT[j][c] = (_Float16)bones[(((size_t)(nIdx * CIN + c) * TT) + t) * BN + j];
    }
    for (int idx = tid; idx < NPAD; idx += 256) sLogits[idx] = 0.f;
    for (int idx = tid; idx < COUT * JN; idx += 256) ((float*)sJOutT)[idx] = 0.f;
    if (tid < BN) { sE0[tid] = edges[tid * 2]; sE1[tid] = edges[tid * 2 + 1]; }
    __syncthreads();

    // ---------------- Phase B: bone path means (bpf) into sPathT -------------
    path_means(wave, lane, bpaths, bmask, blen, Lb, sBT);
    __syncthreads();

    // ---------------- Phase C: attn GEMM  h = gelu(W1*X), logits += W2*h -----
    {
        const int mt = wave & 3;
        const int ntBase = (wave >> 2) * 20;
        v16h A[6];
#pragma unroll
        for (int kt = 0; kt < 6; ++kt) A[kt] = load_a_frag(W1, mt, kt, nl, hi);
        float b1v[8], w2v[8];
#pragma unroll
        for (int r = 0; r < 8; ++r) {
            int m = mt * 16 + r + 8 * hi;
            b1v[r] = b1[m];
            w2v[r] = W2[m];
        }
        for (int nt = ntBase; nt < ntBase + 20; ++nt) {
            int p  = nt * 16 + nl;
            int pc = p < NP ? p : (NP - 1);
            int pi = pc / JN, pj = pc - pi * JN;
            v8f acc = {};
#pragma unroll
            for (int kt = 0; kt < 6; ++kt) {
                // Each lane needs 16 consecutive K values -> one 32B LDS load
                // (2x ds_load_b128), no packing moves.
                v16h Bf;
                if (kt < 2)      Bf = *(const v16h*)&sJT[pi][kt * 32 + hi * 16];
                else if (kt < 4) Bf = *(const v16h*)&sJT[pj][(kt - 2) * 32 + hi * 16];
                else             Bf = *(const v16h*)&sPathT[pc][(kt - 4) * 32 + hi * 16];
                acc = __builtin_amdgcn_wmma_f32_16x16x32_f16(
                        false, A[kt], false, Bf, (short)0, acc, false, false);
            }
            if (p < NP) {
                float lsum = 0.f;
#pragma unroll
                for (int r = 0; r < 8; ++r) {
                    float g = gelu_exact(acc[r] + b1v[r]);
                    lsum += w2v[r] * g;
                }
                atomicAdd(&sLogits[p], lsum);   // ds_add_f32
            }
        }
    }
    __syncthreads();

    // ------- Phase D (tid<25): softmax rows  |  Phase E: jpf into sPathT -----
    if (tid < JN) {
        int i = tid;
        float mx = -3.4e38f;
        for (int j = 0; j < JN; ++j) mx = fmaxf(mx, sLogits[i * JN + j]);
        float s = 0.f;
        float e[JN];
        for (int j = 0; j < JN; ++j) { e[j] = __expf(sLogits[i * JN + j] - mx); s += e[j]; }
        float inv = 1.0f / s;
        for (int j = 0; j < JN; ++j) sAttn[i * JN + j] = e[j] * inv;
    }
    path_means(wave, lane, jpaths, jmask, jlen, Lj, sJT);
    __syncthreads();

    // ---------------- Phase F: value GEMM, attn-weighted accumulation --------
    {
        const int mt = wave & 3;
        const int ntBase = (wave >> 2) * 20;
        v16h A[6];
#pragma unroll
        for (int kt = 0; kt < 6; ++kt) A[kt] = load_a_frag(Wt, mt, kt, nl, hi);
        float btv[8];
#pragma unroll
        for (int r = 0; r < 8; ++r) btv[r] = bt[mt * 16 + r + 8 * hi];
        for (int nt = ntBase; nt < ntBase + 20; ++nt) {
            int p  = nt * 16 + nl;
            int pc = p < NP ? p : (NP - 1);
            int pi = pc / JN, pj = pc - pi * JN;
            v8f acc = {};
#pragma unroll
            for (int kt = 0; kt < 6; ++kt) {
                v16h Bf;
                if (kt < 2)      Bf = *(const v16h*)&sJT[pi][kt * 32 + hi * 16];
                else if (kt < 4) Bf = *(const v16h*)&sJT[pj][(kt - 2) * 32 + hi * 16];
                else             Bf = *(const v16h*)&sPathT[pc][(kt - 4) * 32 + hi * 16];
                acc = __builtin_amdgcn_wmma_f32_16x16x32_f16(
                        false, A[kt], false, Bf, (short)0, acc, false, false);
            }
            if (p < NP) {
                float aw = sAttn[p];
#pragma unroll
                for (int r = 0; r < 8; ++r) {
                    int m = mt * 16 + r + 8 * hi;
                    atomicAdd(&sJOutT[pi][m], (acc[r] + btv[r]) * aw);
                }
            }
        }
    }
    __syncthreads();

    // -------- Phase G: bones GEMM (one 16x16 tile per wave) ------------------
    {
        const int mt  = wave & 3;
        const int nt2 = wave >> 2;           // 0 or 1 -> cols 0-15 / 16-31
        v16h A[6];
#pragma unroll
        for (int kt = 0; kt < 6; ++kt) A[kt] = load_a_frag(Wb, mt, kt, nl, hi);
        float bbv[8];
#pragma unroll
        for (int r = 0; r < 8; ++r) bbv[r] = bb[mt * 16 + r + 8 * hi];

        int col  = nt2 * 16 + nl;
        int colc = col < BN ? col : 0;
        int e0 = sE0[colc], e1 = sE1[colc];
        v8f acc = {};
#pragma unroll
        for (int kt = 0; kt < 6; ++kt) {
            v16h Bf;
            if (kt < 2) {
                Bf = *(const v16h*)&sBT[colc][kt * 32 + hi * 16];
            } else {
                // s2/t2 rows come from the f32 joints_out accumulator:
                // 16 consecutive channels -> 4x float4 LDS loads + cvt.
                const float* srcRow = (kt < 4)
                    ? &sJOutT[e0][(kt - 2) * 32 + hi * 16]
                    : &sJOutT[e1][(kt - 4) * 32 + hi * 16];
                f4 v0 = *(const f4*)(srcRow);
                f4 v1 = *(const f4*)(srcRow + 4);
                f4 v2 = *(const f4*)(srcRow + 8);
                f4 v3 = *(const f4*)(srcRow + 12);
#pragma unroll
                for (int q = 0; q < 4; ++q) {
                    Bf[q]      = (_Float16)v0[q];
                    Bf[q + 4]  = (_Float16)v1[q];
                    Bf[q + 8]  = (_Float16)v2[q];
                    Bf[q + 12] = (_Float16)v3[q];
                }
            }
            acc = __builtin_amdgcn_wmma_f32_16x16x32_f16(
                    false, A[kt], false, Bf, (short)0, acc, false, false);
        }
        if (col < BN) {
            const size_t OUTJ = (size_t)NB * COUT * TT * JN;
#pragma unroll
            for (int r = 0; r < 8; ++r) {
                int m = mt * 16 + r + 8 * hi;
                out[OUTJ + (((size_t)(nIdx * COUT + m) * TT) + t) * BN + col] =
                    acc[r] + bbv[r];
            }
        }
    }

    // -------- Phase H: write joints_out -------------------------------------
    for (int idx = tid; idx < COUT * JN; idx += 256) {
        int c = idx / JN, i = idx - c * JN;
        out[(((size_t)(nIdx * COUT + c) * TT) + t) * JN + i] = sJOutT[i][c];
    }
}

extern "C" void kernel_launch(void* const* d_in, const int* in_sizes, int n_in,
                              void* d_out, int out_size, void* d_ws, size_t ws_size,
                              hipStream_t stream) {
    const float* joints = (const float*)d_in[0];
    const float* bones  = (const float*)d_in[1];
    const float* W1     = (const float*)d_in[2];
    const float* b1     = (const float*)d_in[3];
    const float* W2     = (const float*)d_in[4];
    // d_in[5] = b2 : constant shift before softmax -> no effect on outputs
    const float* Wt     = (const float*)d_in[6];
    const float* bt     = (const float*)d_in[7];
    const float* Wb     = (const float*)d_in[8];
    const float* bb     = (const float*)d_in[9];
    const int*   edges  = (const int*)d_in[10];
    const int*   jpaths = (const int*)d_in[11];
    const float* jmask  = (const float*)d_in[12];
    const float* jlen   = (const float*)d_in[13];
    const int*   bpaths = (const int*)d_in[14];
    const float* bmask  = (const float*)d_in[15];
    const float* blen   = (const float*)d_in[16];
    float* out = (float*)d_out;

    int Lj = in_sizes[11] / (JN * JN);   // path buffer depth
    int Lb = in_sizes[14] / (JN * JN);

    dim3 grid(NB * TT);   // one workgroup per (n, t)
    dim3 block(256);      // 8 wave32s
    spatial_layer_fused<<<grid, block, 0, stream>>>(
        joints, bones, W1, b1, W2, Wt, bt, Wb, bb, edges,
        jpaths, jmask, jlen, bpaths, bmask, blen, out, Lj, Lb);
}
